// LandmarkGAT_58102317580659
// MI455X (gfx1250) — matrix operand order repaired and checked by reading, not compiled, not run on hardware
//
#include <hip/hip_runtime.h>
#include <cstdint>
#include <cstddef>

typedef __attribute__((ext_vector_type(16))) __bf16 v16bf;
typedef __attribute__((ext_vector_type(8)))  float  v8f;

#define NEG_SLOPE 0.2f

// ---------- helpers ----------
__device__ __forceinline__ unsigned short f2bf(float f) {
    unsigned u = __float_as_uint(f);
    unsigned r = u + 0x7FFFu + ((u >> 16) & 1u);   // round-to-nearest-even
    return (unsigned short)(r >> 16);
}
// monotonic float -> ordered uint key (for atomicMax-based segment max)
__device__ __forceinline__ unsigned fkey(float f) {
    unsigned u = __float_as_uint(f);
    return (u & 0x80000000u) ? ~u : (u | 0x80000000u);
}
__device__ __forceinline__ float fkey_inv(unsigned k) {
    return __uint_as_float((k & 0x80000000u) ? (k ^ 0x80000000u) : ~k);
}

// ---------- conversion / utility kernels ----------
__global__ void k_f32_to_bf16(const float* __restrict__ a, unsigned short* __restrict__ o, int count) {
    int i = blockIdx.x * blockDim.x + threadIdx.x;
    if (i < count) o[i] = f2bf(a[i]);
}
// W[K,N] row-major -> Wt[N,K] bf16 (so B-fragment K runs are contiguous)
__global__ void k_w_transpose_bf16(const float* __restrict__ W, unsigned short* __restrict__ o, int K, int N) {
    int i = blockIdx.x * blockDim.x + threadIdx.x;
    if (i >= K * N) return;
    int k = i / N, nn = i % N;
    o[(size_t)nn * K + k] = f2bf(W[(size_t)k * N + nn]);
}
__global__ void k_zero_u32(unsigned* __restrict__ p, long count) {
    long i = (long)blockIdx.x * blockDim.x + threadIdx.x;
    if (i < count) p[i] = 0u;
}

// ---------- WMMA GEMM: C[M,N] = A[M,K](bf16) x Wt[N,K]^T(bf16), f32 accum ----------
// Each wave computes a 16 x (16*NT) strip: one A fragment feeds NT WMMAs per k-step.
template <int NT>
__global__ void k_gemm_wmma(const unsigned short* __restrict__ A,
                            const unsigned short* __restrict__ Wt,
                            float* __restrict__ C, int M, int K, int N) {
    int wave = (int)((blockIdx.x * blockDim.x + threadIdx.x) >> 5);
    int lane = threadIdx.x & 31;
    int groups_n = N / (16 * NT);
    int tiles_m  = M >> 4;
    if (wave >= tiles_m * groups_n) return;
    int m0 = (wave / groups_n) << 4;
    int n0 = (wave % groups_n) * (16 * NT);
    int half = lane >> 4;          // 0: lanes 0-15, 1: lanes 16-31
    int l15  = lane & 15;

    int arow  = m0 + l15;          // A fragment: M = lane&15
    int abase = half * 8;          // K base offset (0 or 8)
    int bbase = half * 8;          // K base offset (0 or 8)

    v8f acc[NT] = {};
    union Frag { unsigned short s[16]; uint4 q[2]; v16bf v; };

    for (int k0 = 0; k0 < K; k0 += 32) {
        Frag af;
        const unsigned short* ap = A + (size_t)arow * K + k0 + abase;
        // A: elems 0..7 -> K = base+e ; elems 8..15 -> K = base+16+(e-8)
        af.q[0] = *reinterpret_cast<const uint4*>(ap);
        af.q[1] = *reinterpret_cast<const uint4*>(ap + 16);
#pragma unroll
        for (int t = 0; t < NT; ++t) {
            Frag bf;
            const unsigned short* bp = Wt + (size_t)(n0 + t * 16 + l15) * K + k0 + bbase;
            bf.q[0] = *reinterpret_cast<const uint4*>(bp);
            bf.q[1] = *reinterpret_cast<const uint4*>(bp + 16);
            acc[t] = __builtin_amdgcn_wmma_f32_16x16x32_bf16(false, af.v, false, bf.v,
                                                             (short)0, acc[t], false, false);
        }
    }
#pragma unroll
    for (int t = 0; t < NT; ++t) {
        int ncol = n0 + t * 16 + l15;
#pragma unroll
        for (int r = 0; r < 8; ++r) {
            int m = m0 + r + half * 8;
            C[(size_t)m * N + ncol] = acc[t][r];
        }
    }
}

// ---------- attention coefficients: alpha_s / alpha_d ----------
__global__ void k_alpha(const float* __restrict__ h, const float* __restrict__ a_src,
                        const float* __restrict__ a_dst, float* __restrict__ as_,
                        float* __restrict__ ad_, int n, int H, int d) {
    int idx = blockIdx.x * blockDim.x + threadIdx.x;
    if (idx >= n * H) return;
    int i = idx / H, hh = idx % H;
    const float* hp = h + (size_t)i * H * d + (size_t)hh * d;
    float s = 0.f, t = 0.f;
    for (int j = 0; j < d; ++j) {
        float v = hp[j];
        s += v * a_src[hh * d + j];
        t += v * a_dst[hh * d + j];
    }
    as_[idx] = s;
    ad_[idx] = t;
}

// ---------- edge pass 1: segment max of leaky-relu'd logits ----------
__global__ void k_edge_max(const int* __restrict__ ei, int E, int n, int H,
                           const float* __restrict__ as_, const float* __restrict__ ad_,
                           unsigned* __restrict__ mkey) {
    int idx = blockIdx.x * blockDim.x + threadIdx.x;
    int ET = E + n;
    int e = idx / H, hh = idx % H;
    if (e >= ET) return;
    int s, t;
    if (e < E) { s = ei[e]; t = ei[E + e]; } else { s = t = e - E; }
    float v = as_[s * H + hh] + ad_[t * H + hh];
    v = v > 0.f ? v : NEG_SLOPE * v;
    atomicMax(&mkey[t * H + hh], fkey(v));
}

// ---------- edge pass 2: one thread per (edge, head, feature) -> coalesced ----------
__global__ void k_edge_acc(const int* __restrict__ ei, int E, int n, int H, int d,
                           const float* __restrict__ as_, const float* __restrict__ ad_,
                           const unsigned* __restrict__ mkey, const float* __restrict__ h,
                           float* __restrict__ denom, float* __restrict__ acc) {
    long idx = (long)blockIdx.x * blockDim.x + threadIdx.x;
    int HD = H * d;
    long total = (long)(E + n) * HD;
    if (idx >= total) return;
    int e   = (int)(idx / HD);
    int rem = (int)(idx % HD);
    int hh  = rem / d;
    int j   = rem % d;
    int s, t;
    if (e < E) { s = ei[e]; t = ei[E + e]; } else { s = t = e - E; }
    float v = as_[s * H + hh] + ad_[t * H + hh];
    v = v > 0.f ? v : NEG_SLOPE * v;
    float m = fkey_inv(mkey[t * H + hh]);
    float w = __expf(v - m);
    if (j == 0) atomicAdd(&denom[t * H + hh], w);
    atomicAdd(&acc[(size_t)t * HD + rem], w * h[(size_t)s * HD + rem]);
}

// ---------- normalize by denom and add bias (in place on acc) ----------
__global__ void k_finalize(float* __restrict__ acc, const float* __restrict__ denom,
                           const float* __restrict__ bias, int n, int H, int d) {
    int idx = blockIdx.x * blockDim.x + threadIdx.x;
    int HD = H * d;
    if (idx >= n * HD) return;
    int i = idx / HD, c = idx % HD, hh = c / d;
    acc[idx] = acc[idx] / (denom[i * H + hh] + 1e-16f) + bias[c];
}

// ---------- batch-norm: row-parallel coalesced partial sums (blockDim == C) ----------
__global__ void k_bn_partial(const float* __restrict__ x, float* __restrict__ stats, int n, int C) {
    int c = threadIdx.x;               // blockDim.x == C
    float s = 0.f, q = 0.f;
    for (int r = blockIdx.x; r < n; r += gridDim.x) {
        float v = x[(size_t)r * C + c];
        s += v; q += v * v;
    }
    atomicAdd(&stats[c], s);
    atomicAdd(&stats[C + c], q);
}
__global__ void k_bn_final(float* __restrict__ stats, int n, int C) {
    int c = blockIdx.x * blockDim.x + threadIdx.x;
    if (c >= C) return;
    float mean = stats[c] / (float)n;
    float var  = stats[C + c] / (float)n - mean * mean;
    stats[c]     = mean;
    stats[C + c] = rsqrtf(var + 1e-5f);
}

// ---------- BN apply + SELU (in place) ----------
__global__ void k_bn_selu(float* __restrict__ x, const float* __restrict__ stats,
                          const float* __restrict__ g, const float* __restrict__ b, int n, int C) {
    int idx = blockIdx.x * blockDim.x + threadIdx.x;
    if (idx >= n * C) return;
    int c = idx % C;
    float v = (x[idx] - stats[c]) * stats[C + c] * g[c] + b[c];
    const float sc = 1.0507009873554805f, al = 1.6732632423543772f;
    x[idx] = v > 0.f ? sc * v : sc * al * (expf(v) - 1.f);
}

// ---------- pooling scores ----------
__global__ void k_scores(const float* __restrict__ h3, const float* __restrict__ pw,
                         const float* __restrict__ pb, float* __restrict__ sc, int n, int C) {
    int i = blockIdx.x * blockDim.x + threadIdx.x;
    if (i >= n) return;
    float s = pb[0];
    for (int c = 0; c < C; ++c) s += h3[(size_t)i * C + c] * pw[c];
    sc[i] = 1.f / (1.f + expf(-s));
}

// ---------- stable top-k select + gate + gather (one block per batch) ----------
__global__ void k_topk(const float* __restrict__ h3, const float* __restrict__ sc,
                       float* __restrict__ out, int Nn, int k, int C) {
    __shared__ float s[512];
    int b = blockIdx.x;
    for (int j = threadIdx.x; j < Nn; j += blockDim.x) s[j] = sc[b * Nn + j];
    __syncthreads();
    for (int j = threadIdx.x; j < Nn; j += blockDim.x) {
        float sj = s[j];
        int r = 0;
        for (int i = 0; i < Nn; ++i) {
            float si = s[i];
            r += (si > sj) || (si == sj && i < j);
        }
        if (r < k) {
            const float* hp = h3 + ((size_t)(b * Nn + j)) * C;
            float* op = out + ((size_t)b * k + r) * C;
            for (int c = 0; c < C; ++c) op[c] = hp[c] * sj;
        }
    }
}

// ---------- host orchestration ----------
extern "C" void kernel_launch(void* const* d_in, const int* in_sizes, int n_in,
                              void* d_out, int out_size, void* d_ws, size_t ws_size,
                              hipStream_t stream) {
    (void)n_in; (void)out_size; (void)ws_size;
    const float* x    = (const float*)d_in[0];
    const int*   ei   = (const int*)  d_in[1];
    const float* W1   = (const float*)d_in[2];
    const float* aS1  = (const float*)d_in[3];
    const float* aD1  = (const float*)d_in[4];
    const float* b1   = (const float*)d_in[5];
    const float* W2   = (const float*)d_in[6];
    const float* aS2  = (const float*)d_in[7];
    const float* aD2  = (const float*)d_in[8];
    const float* b2   = (const float*)d_in[9];
    const float* W3   = (const float*)d_in[10];
    const float* aS3  = (const float*)d_in[11];
    const float* aD3  = (const float*)d_in[12];
    const float* b3   = (const float*)d_in[13];
    const float* bn1g = (const float*)d_in[14];
    const float* bn1b = (const float*)d_in[15];
    const float* bn2g = (const float*)d_in[16];
    const float* bn2b = (const float*)d_in[17];
    const float* pw   = (const float*)d_in[18];
    const float* pb   = (const float*)d_in[19];

    const int Bsz = 128, Nn = 478, n = Bsz * Nn;   // 61184 nodes (multiple of 16)
    const int E = in_sizes[1] / 2;
    const int kSel = Nn / 2;                       // 239

    // workspace carve
    char* ws = (char*)d_ws;
    size_t off = 0;
    auto carve = [&](size_t bytes) -> void* {
        bytes = (bytes + 255) & ~(size_t)255;
        void* p = ws + off; off += bytes; return p;
    };
    float*          bufA  = (float*)         carve((size_t)n * 256 * 4);
    float*          bufB  = (float*)         carve((size_t)n * 256 * 4);
    unsigned short* abf   = (unsigned short*)carve((size_t)n * 256 * 2);
    unsigned short* wbf   = (unsigned short*)carve((size_t)256 * 256 * 2);
    float*          as_   = (float*)         carve((size_t)n * 8 * 4);
    float*          ad_   = (float*)         carve((size_t)n * 8 * 4);
    unsigned*       mkey  = (unsigned*)      carve((size_t)n * 8 * 4);
    float*          denom = (float*)         carve((size_t)n * 8 * 4);
    float*          stats = (float*)         carve((size_t)2 * 256 * 4);
    float*          score = (float*)         carve((size_t)n * 4);

    auto cdiv = [](long a, long b) -> int { return (int)((a + b - 1) / b); };

    // One GAT layer: hin[n,K] -> hbuf = hin@W (WMMA), attention-aggregate into accbuf,
    // finalize accbuf (normalize + bias). hin may alias accbuf (consumed before zeroing).
    auto gat_layer = [&](const float* hin, int K, int H, int d,
                         const float* W, const float* aS, const float* aD,
                         const float* bias, float* hbuf, float* accbuf) {
        int HD = H * d;
        k_f32_to_bf16<<<cdiv((long)n * K, 256), 256, 0, stream>>>(hin, abf, n * K);
        k_w_transpose_bf16<<<cdiv((long)K * HD, 256), 256, 0, stream>>>(W, wbf, K, HD);
        if (HD % 64 == 0) {
            int waves = (n / 16) * (HD / 64);
            k_gemm_wmma<4><<<cdiv((long)waves * 32, 256), 256, 0, stream>>>(abf, wbf, hbuf, n, K, HD);
        } else {
            int waves = (n / 16) * (HD / 16);
            k_gemm_wmma<1><<<cdiv((long)waves * 32, 256), 256, 0, stream>>>(abf, wbf, hbuf, n, K, HD);
        }
        k_alpha<<<cdiv((long)n * H, 256), 256, 0, stream>>>(hbuf, aS, aD, as_, ad_, n, H, d);
        k_zero_u32<<<cdiv((long)n * H, 256), 256, 0, stream>>>(mkey, (long)n * H);
        k_zero_u32<<<cdiv((long)n * H, 256), 256, 0, stream>>>((unsigned*)denom, (long)n * H);
        k_zero_u32<<<cdiv((long)n * HD, 256), 256, 0, stream>>>((unsigned*)accbuf, (long)n * HD);
        int ET = E + n;
        k_edge_max<<<cdiv((long)ET * H, 256), 256, 0, stream>>>(ei, E, n, H, as_, ad_, mkey);
        k_edge_acc<<<cdiv((long)ET * HD, 256), 256, 0, stream>>>(ei, E, n, H, d, as_, ad_, mkey,
                                                                 hbuf, denom, accbuf);
        k_finalize<<<cdiv((long)n * HD, 256), 256, 0, stream>>>(accbuf, denom, bias, n, H, d);
    };

    auto batchnorm_selu = [&](float* buf, int C, const float* g, const float* b) {
        k_zero_u32<<<cdiv(2L * C, 256), 256, 0, stream>>>((unsigned*)stats, 2L * C);
        k_bn_partial<<<256, C, 0, stream>>>(buf, stats, n, C);
        k_bn_final<<<cdiv((long)C, 256), 256, 0, stream>>>(stats, n, C);
        k_bn_selu<<<cdiv((long)n * C, 256), 256, 0, stream>>>(buf, stats, g, b, n, C);
    };

    // Layer 1: x[n,64] -> bufB[n,256], then BN+SELU
    gat_layer(x, 64, 8, 32, W1, aS1, aD1, b1, bufA, bufB);
    batchnorm_selu(bufB, 256, bn1g, bn1b);

    // Layer 2: bufB[n,256] -> bufB[n,128], then BN+SELU
    gat_layer(bufB, 256, 8, 16, W2, aS2, aD2, b2, bufA, bufB);
    batchnorm_selu(bufB, 128, bn2g, bn2b);

    // Layer 3: bufB[n,128] -> bufB[n,16] (1 head, mean over 1 head == identity)
    gat_layer(bufB, 128, 1, 16, W3, aS3, aD3, b3, bufA, bufB);

    // Pooling: scores, stable top-k (descending, ties -> lower index), gate, gather
    k_scores<<<cdiv((long)n, 256), 256, 0, stream>>>(bufB, pw, pb, score, n, 16);
    k_topk<<<Bsz, 512, 0, stream>>>(bufB, score, (float*)d_out, Nn, kSel, 16);
}
